// GraphFusionNet_50869592654340
// MI455X (gfx1250) — matrix operand order repaired and checked by reading, not compiled, not run on hardware
//
#include <hip/hip_runtime.h>
#include <hip/hip_bf16.h>

#define N_   16
#define E_   512
#define IND  1024
#define H_   8
#define T_   2
#define O_   128
#define HT_  (H_*T_)
#define NE_  (N_*E_)          // 8192 rows in the big GEMM
#define NEGV (-9.0e15f)

typedef __attribute__((ext_vector_type(16))) __bf16       v16bf;
typedef __attribute__((ext_vector_type(8)))  float        v8f;
typedef __attribute__((ext_vector_type(4)))  unsigned int u32x4;
typedef __attribute__((ext_vector_type(8)))  int          i32x8;
typedef __attribute__((ext_vector_type(4)))  int          i32x4;

union ABFrag { v16bf v; u32x4 q[2]; };
union Pack8  { unsigned short s[8]; u32x4 q; };

static __device__ __forceinline__ unsigned short f2bf(float f) {
  unsigned int u = __float_as_uint(f);
  u += 0x7FFFu + ((u >> 16) & 1u);     // round-to-nearest-even
  return (unsigned short)(u >> 16);
}

// TDM: DMA a 32-row x 64-byte tile (row stride 2048B) from global into LDS,
// inserting 16B of padding after every 64B so LDS rows land at stride 80B
// (bank-conflict-free for the WMMA fragment reads).
static __device__ __forceinline__ void tdm_load_tile(unsigned lds_off, const void* gptr) {
  unsigned long long ga = (unsigned long long)(uintptr_t)gptr;
  u32x4 g0;
  g0[0] = 1u;                                                 // count=1, user mode
  g0[1] = lds_off;                                            // lds_addr
  g0[2] = (unsigned)(ga & 0xFFFFFFFFu);                       // global_addr[31:0]
  g0[3] = (unsigned)((ga >> 32) & 0x01FFFFFFu) | (2u << 30);  // addr[56:32] | type=2
  i32x8 g1;
  g1[0] = (int)((0u << 16)        // data_size = 1 byte
              | (1u << 20)        // pad_enable
              | (3u << 22)        // pad_interval: 16 DWORDs (=64B) before padding
              | (3u << 25));      // pad_amount: 4 DWORDs (=16B)
  g1[1] = (int)(64u << 16);       // tensor_dim0 = 64 bytes
  g1[2] = (int)(32u << 16);       // tensor_dim1 = 32 rows
  g1[3] = (int)(64u << 16);       // tile_dim0 = 64 bytes
  g1[4] = (int)32u;               // tile_dim1 = 32 rows (tile_dim2 = 0)
  g1[5] = (int)2048u;             // tensor_dim0_stride = 2048 bytes
  g1[6] = 0; g1[7] = 0;
  i32x4 z4 = {};
  i32x8 z8 = {};
  __builtin_amdgcn_tensor_load_to_lds(g0, g1, z4, z4, z8, 0);  // 6-arg (clang-23) form
}

// ---------------- fp32 -> bf16 convert of input_state (2 elems/thread) -----
__global__ void k_cvt_in(const float* __restrict__ in, unsigned int* __restrict__ out, int n2) {
  int i = blockIdx.x * blockDim.x + threadIdx.x;
  if (i < n2)
    out[i] = (unsigned int)f2bf(in[2*i]) | ((unsigned int)f2bf(in[2*i+1]) << 16);
}

// ---------------- W (H,T,D,O) fp32 -> Wt[ht][o][d] bf16 (K-contiguous B) ---
__global__ void k_cvt_w(const float* __restrict__ W, unsigned short* __restrict__ Wt) {
  int idx = blockIdx.x * blockDim.x + threadIdx.x;   // (ht*128+o)*1024 + d
  int d  = idx & (IND - 1);
  int o  = (idx >> 10) & (O_ - 1);
  int ht = idx >> 17;
  Wt[idx] = f2bf(W[(ht * IND + d) * O_ + o]);
}

// ---------------- adj[n][i][j] -> adjT[n][j][i] (LDS tiled transpose) ------
__global__ void k_adjT(const int* __restrict__ adj, int* __restrict__ adjT) {
  __shared__ int tile[32][33];
  int n  = blockIdx.z;
  int i0 = blockIdx.y * 32, j0 = blockIdx.x * 32;
  int tx = threadIdx.x, ty = threadIdx.y;
  tile[ty][tx] = adj[(n * E_ + i0 + ty) * E_ + j0 + tx];
  __syncthreads();
  adjT[(n * E_ + j0 + ty) * E_ + i0 + tx] = tile[tx][ty];
}

// ---------------- gate MLP: v1 = sigmoid(..)[:O]*a_src, v2 = [O:]*a_dst ----
__global__ __launch_bounds__(256) void k_gate(
    const float* __restrict__ qv,  const float* __restrict__ qW1,
    const float* __restrict__ qW2, const float* __restrict__ a_src,
    const float* __restrict__ a_dst, float* __restrict__ v1, float* __restrict__ v2) {
  __shared__ float q[IND];
  __shared__ float tmp[2 * O_];
  int ht = blockIdx.x >> 4;
  int n  = blockIdx.x & 15;
  int c  = threadIdx.x;                       // 0..255
  for (int d = c; d < IND; d += 256) q[d] = qv[n * IND + d];
  __syncthreads();
  float z = 0.f;
  for (int d = 0; d < IND; ++d) z += q[d] * qW1[(ht * IND + d) * (2 * O_) + c];
  tmp[c] = z > 0.f ? z : 0.f;                 // relu
  __syncthreads();
  float z2 = 0.f;
  for (int d = 0; d < 2 * O_; ++d) z2 += tmp[d] * qW2[(ht * 2 * O_ + d) * (2 * O_) + c];
  float g = 1.f / (1.f + __expf(-z2));        // sigmoid
  if (c < O_) v1[(ht * N_ + n) * O_ + c]        = g * a_src[ht * O_ + c];
  else        v2[(ht * N_ + n) * O_ + (c - O_)] = g * a_dst[ht * O_ + (c - O_)];
}

// ---------------- GEMM1: h = in @ W  (bf16 WMMA, TDM-staged A, fused epilogue)
__global__ __launch_bounds__(256) void k_gemm1(
    const unsigned short* __restrict__ inb,   // [NE][IND] bf16
    const unsigned short* __restrict__ Wt,    // [HT*O][IND] bf16
    const float* __restrict__ v1, const float* __restrict__ v2,
    float* __restrict__ s1, float* __restrict__ s2,
    unsigned short* __restrict__ hl)          // [H][N][O][E] bf16 (h_last^T)
{
  const int ht  = blockIdx.y;
  const int m0  = blockIdx.x * 32;            // global row (n,e) base
  const int n   = m0 >> 9;
  const int tid = threadIdx.x;
  const int w = tid >> 5, lane = tid & 31, lg = lane >> 4, lm = lane & 15;

  __shared__ unsigned short As[2][32 * 40];   // double-buffered 32x32 tiles, stride 40
  __shared__ float s1s[32], s2s[32];
  if (tid < 32) { s1s[tid] = 0.f; s2s[tid] = 0.f; }

  v8f acc0 = {}; v8f acc1 = {};
  const int ocol = w * 16 + lm;               // this wave's 16 columns of O
  const unsigned short* Bp = Wt + (ht * O_ + ocol) * IND;
  const unsigned short* Ap = inb + (size_t)m0 * IND;
  const unsigned lds0 = (unsigned)(uintptr_t)(&As[0][0]);
  const unsigned lds1 = (unsigned)(uintptr_t)(&As[1][0]);

  if (w == 0) {                               // prologue: DMA tile 0
    tdm_load_tile(lds0, Ap);
    __builtin_amdgcn_s_wait_tensorcnt(0);
  }
  __syncthreads();

  for (int k0 = 0; k0 < IND; k0 += 32) {
    const int buf = (k0 >> 5) & 1;
    if (w == 0 && k0 + 32 < IND)              // prefetch next tile into other buffer
      tdm_load_tile(buf ? lds0 : lds1, Ap + k0 + 32);

    const unsigned short* Asb = &As[buf][0];
    ABFrag a0, a1, b;
    a0.q[0] = *(const u32x4*)(Asb + lm * 40 + 8 * lg);
    a0.q[1] = *(const u32x4*)(Asb + lm * 40 + 16 + 8 * lg);
    a1.q[0] = *(const u32x4*)(Asb + (16 + lm) * 40 + 8 * lg);
    a1.q[1] = *(const u32x4*)(Asb + (16 + lm) * 40 + 16 + 8 * lg);
    b.q[0]  = *(const u32x4*)(Bp + k0 + 8 * lg);
    b.q[1]  = *(const u32x4*)(Bp + k0 + 16 + 8 * lg);
    acc0 = __builtin_amdgcn_wmma_f32_16x16x32_bf16(false, a0.v, false, b.v, (short)0, acc0, false, false);
    acc1 = __builtin_amdgcn_wmma_f32_16x16x32_bf16(false, a1.v, false, b.v, (short)0, acc1, false, false);

    if (w == 0)
      __builtin_amdgcn_s_wait_tensorcnt(0);   // next tile resident before barrier
    __syncthreads();
  }

  // fused s1/s2: dot of each h-row with v1/v2 (reduce cols via LDS atomics)
  const float f1 = v1[(ht * N_ + n) * O_ + ocol];
  const float f2 = v2[(ht * N_ + n) * O_ + ocol];
#pragma unroll
  for (int r = 0; r < 8; ++r) {               // acc0 row = r+8*lg; acc1 row = 16+r+8*lg
    atomicAdd(&s1s[r + 8 * lg],      acc0[r] * f1);
    atomicAdd(&s2s[r + 8 * lg],      acc0[r] * f2);
    atomicAdd(&s1s[16 + r + 8 * lg], acc1[r] * f1);
    atomicAdd(&s2s[16 + r + 8 * lg], acc1[r] * f2);
  }
  __syncthreads();
  if (tid < 32) {
    s1[ht * NE_ + m0 + tid] = s1s[tid];
    s2[ht * NE_ + m0 + tid] = s2s[tid];
  }

  if (ht & 1) {                               // t == T-1: store h_last as [o][e] bf16
    const int hh = ht >> 1;
    const int e0 = (m0 & (E_ - 1)) + 8 * lg;
    Pack8 p0, p1;
#pragma unroll
    for (int r = 0; r < 8; ++r) { p0.s[r] = f2bf(acc0[r]); p1.s[r] = f2bf(acc1[r]); }
    unsigned short* hp = hl + (size_t)((hh * N_ + n) * O_ + ocol) * E_;
    *(u32x4*)(hp + e0)      = p0.q;
    *(u32x4*)(hp + e0 + 16) = p1.q;
  }
}

// ---------------- scores + online softmax -> coefsT[h][n][j][i] bf16 -------
__global__ __launch_bounds__(512) void k_softmax(
    const float* __restrict__ s1, const float* __restrict__ s2,
    const int* __restrict__ adjT, unsigned short* __restrict__ coefsT)
{
  const int n = blockIdx.x, h = blockIdx.y;
  const int i = threadIdx.x;                  // row of score matrix
  __shared__ float s2s[2][E_];
  s2s[0][i] = s2[(h * 2 + 0) * NE_ + n * E_ + i];
  s2s[1][i] = s2[(h * 2 + 1) * NE_ + n * E_ + i];
  const float a1v = s1[(h * 2 + 0) * NE_ + n * E_ + i];
  const float a2v = s1[(h * 2 + 1) * NE_ + n * E_ + i];
  __syncthreads();
  const int* ap = adjT + (size_t)n * E_ * E_;
  float mx = NEGV, sum = 0.f;
  for (int j = 0; j < E_; ++j) {              // pass 1: online max + sum
    int a = ap[(size_t)j * E_ + i];
    float s = NEGV;
    if (a == 1)      { s = a1v + s2s[0][j]; s = s > 0.f ? s : 0.2f * s; }
    else if (a == 2) { s = a2v + s2s[1][j]; s = s > 0.f ? s : 0.2f * s; }
    if (s > mx) { sum = sum * __expf(mx - s) + 1.f; mx = s; }
    else        { sum += __expf(s - mx); }
  }
  const float inv = 1.f / sum;
  unsigned short* cp = coefsT + (size_t)(h * N_ + n) * E_ * E_;
  for (int j = 0; j < E_; ++j) {              // pass 2: write transposed coefs
    int a = ap[(size_t)j * E_ + i];
    float s = NEGV;
    if (a == 1)      { s = a1v + s2s[0][j]; s = s > 0.f ? s : 0.2f * s; }
    else if (a == 2) { s = a2v + s2s[1][j]; s = s > 0.f ? s : 0.2f * s; }
    cp[(size_t)j * E_ + i] = f2bf(__expf(s - mx) * inv);
  }
}

// ---------------- GEMM2: out = coefs^T @ h_last, fused ReLU + layout -------
__global__ __launch_bounds__(256) void k_gemm2(
    const unsigned short* __restrict__ coefsT, // [H][N][E(j)][E(i)] bf16
    const unsigned short* __restrict__ hl,     // [H][N][O][E(i)]    bf16
    float* __restrict__ out)                   // [N][E][H*O]
{
  const int jt = blockIdx.x, n = blockIdx.y, h = blockIdx.z;
  const int j0 = jt * 32;
  const int tid = threadIdx.x;
  const int w = tid >> 5, lane = tid & 31, lg = lane >> 4, lm = lane & 15;
  const unsigned short* Ab = coefsT + (size_t)(h * N_ + n) * E_ * E_;
  const unsigned short* Bb = hl + (size_t)((h * N_ + n) * O_ + w * 16 + lm) * E_;
  v8f acc0 = {}, acc1 = {};
  for (int k0 = 0; k0 < E_; k0 += 32) {
    ABFrag a0, a1, b;
    const unsigned short* ar0 = Ab + (size_t)(j0 + lm) * E_ + k0;
    const unsigned short* ar1 = Ab + (size_t)(j0 + 16 + lm) * E_ + k0;
    a0.q[0] = *(const u32x4*)(ar0 + 8 * lg);
    a0.q[1] = *(const u32x4*)(ar0 + 16 + 8 * lg);
    a1.q[0] = *(const u32x4*)(ar1 + 8 * lg);
    a1.q[1] = *(const u32x4*)(ar1 + 16 + 8 * lg);
    b.q[0]  = *(const u32x4*)(Bb + k0 + 8 * lg);
    b.q[1]  = *(const u32x4*)(Bb + k0 + 16 + 8 * lg);
    acc0 = __builtin_amdgcn_wmma_f32_16x16x32_bf16(false, a0.v, false, b.v, (short)0, acc0, false, false);
    acc1 = __builtin_amdgcn_wmma_f32_16x16x32_bf16(false, a1.v, false, b.v, (short)0, acc1, false, false);
  }
  const int col = h * O_ + w * 16 + lm;
#pragma unroll
  for (int r = 0; r < 8; ++r) {
    int j  = j0 + r + 8 * lg;
    int j2 = j0 + 16 + r + 8 * lg;
    float v  = acc0[r];
    float v2 = acc1[r];
    out[(size_t)(n * E_ + j)  * (H_ * O_) + col] = v  > 0.f ? v  : 0.f;
    out[(size_t)(n * E_ + j2) * (H_ * O_) + col] = v2 > 0.f ? v2 : 0.f;
  }
}

extern "C" void kernel_launch(void* const* d_in, const int* in_sizes, int n_in,
                              void* d_out, int out_size, void* d_ws, size_t ws_size,
                              hipStream_t stream) {
  (void)in_sizes; (void)n_in; (void)out_size; (void)ws_size;
  const float* input_state = (const float*)d_in[0];
  const int*   adj         = (const int*)d_in[1];
  /* d_in[2] entity_mask: all-ones, unused by reference */
  const float* query_vec   = (const float*)d_in[3];
  const float* W           = (const float*)d_in[4];
  const float* a_src       = (const float*)d_in[5];
  const float* a_dst       = (const float*)d_in[6];
  const float* qW1         = (const float*)d_in[7];
  const float* qW2         = (const float*)d_in[8];
  float* out = (float*)d_out;

  char* ws = (char*)d_ws;
  size_t off = 0;
  auto alloc = [&](size_t bytes) {
    char* p = ws + off;
    off = (off + bytes + 255) & ~(size_t)255;
    return p;
  };
  unsigned short* inb    = (unsigned short*)alloc((size_t)NE_ * IND * 2);        // 16.8 MB
  unsigned short* Wt     = (unsigned short*)alloc((size_t)HT_ * O_ * IND * 2);   //  4.2 MB
  float*          v1     = (float*)alloc((size_t)HT_ * N_ * O_ * 4);
  float*          v2     = (float*)alloc((size_t)HT_ * N_ * O_ * 4);
  float*          s1     = (float*)alloc((size_t)HT_ * NE_ * 4);
  float*          s2     = (float*)alloc((size_t)HT_ * NE_ * 4);
  unsigned short* hl     = (unsigned short*)alloc((size_t)H_ * N_ * O_ * E_ * 2); // 16.8 MB
  int*            adjT   = (int*)alloc((size_t)N_ * E_ * E_ * 4);                 // 16.8 MB
  unsigned short* coefsT = (unsigned short*)alloc((size_t)H_ * N_ * E_ * E_ * 2); // 67.1 MB

  k_cvt_in<<<dim3((NE_ * IND / 2) / 256), 256, 0, stream>>>(input_state, (unsigned int*)inb, NE_ * IND / 2);
  k_cvt_w<<<dim3((HT_ * O_ * IND) / 256), 256, 0, stream>>>(W, Wt);
  k_adjT<<<dim3(E_ / 32, E_ / 32, N_), dim3(32, 32), 0, stream>>>(adj, adjT);
  k_gate<<<dim3(HT_ * N_), 256, 0, stream>>>(query_vec, qW1, qW2, a_src, a_dst, v1, v2);
  k_gemm1<<<dim3(NE_ / 32, HT_), 256, 0, stream>>>(inb, Wt, v1, v2, s1, s2, hl);
  k_softmax<<<dim3(N_, H_), E_, 0, stream>>>(s1, s2, adjT, coefsT);
  k_gemm2<<<dim3(E_ / 32, N_, H_), 256, 0, stream>>>(coefsT, hl, out);
}
